// SpikeFP32Adder_43860206027301
// MI455X (gfx1250) — matrix operand order repaired and checked by reading, not compiled, not run on hardware
//
#include <hip/hip_runtime.h>
#include <stdint.h>

typedef __attribute__((ext_vector_type(16))) _Float16 v16h;
typedef __attribute__((ext_vector_type(8)))  float    v8f;
typedef __attribute__((ext_vector_type(4)))  float    v4f;

// Each wave processes a tile of 16 batch rows.
//  - Pack: v_wmma_f32_16x16x32_f16 with a powers-of-two weight matrix gives
//    hi/lo 16-bit halves of each row's packed word, exactly (integer-exact
//    f16 products, f32 accumulation < 2^16).
//  - Add:  one hardware v_add_f32 (IEEE RNE, denormals enabled -> matches the
//    emulated circuit), NaN canonicalized to the circuit's 0x7FFFFFFF.
//  - Unpack: each lane writes 16 of its row's 32 output bits (lane group 0:
//    cols 0-15, group 1: cols 16-31) -> all 32 lanes busy on stores.
__global__ __launch_bounds__(256) void fp32adder_wmma_kernel(
    const float* __restrict__ A, const float* __restrict__ B,
    float* __restrict__ Out, int ntiles)
{
  const int lane = threadIdx.x & 31;
  const int tile = blockIdx.x * 8 + (threadIdx.x >> 5);   // 8 waves / block
  if (tile >= ntiles) return;                             // wave-uniform guard

  const int m = lane & 15;   // batch row within tile / WMMA row & column index
  const int g = lane >> 4;   // K half: 0 -> K=0..15, 1 -> K=16..31

  // ---- constant weight operand: WMMA A-matrix, 16x32 f16 ------------------
  // A-layout: lane m (<16) holds row m, K in {0..15};
  // lane m+16 holds row m, K in {16..31}; element e -> K = 16*g + e.
  // Rows 0 and 8: hi weights 2^(15-k) for k<16.  Rows 1 and 9: lo weights
  // 2^(31-k) for k>=16.  Rows 8/9 duplicate 0/1 so lanes 16-31 of the
  // accumulator (M=8,9) also receive hi/lo of their row.
  v16h w;
#pragma unroll
  for (int e = 0; e < 16; ++e) {
    const int k = g * 16 + e;
    float wv = 0.0f;
    if ((m == 0 || m == 8) && k < 16)  wv = (float)(1u << (15 - k));
    if ((m == 1 || m == 9) && k >= 16) wv = (float)(1u << (31 - k));
    w[e] = (_Float16)wv;
  }

  // ---- bit operand: WMMA B-matrix, 32x16 f16 ------------------------------
  // B-layout: lane n (<16) holds column n, K=0..15; lane n+16 holds column n,
  // K=16..31; elements sequential in K.  Column n = batch row n of the tile,
  // so each lane just loads 16 contiguous floats (64 B) and converts.
  const size_t rowBase = ((size_t)tile * 16 + (size_t)m) * 32 + (size_t)g * 16;
  const float* pa = A + rowBase;
  const float* pb = B + rowBase;
  v16h ba, bb;
#pragma unroll
  for (int q = 0; q < 4; ++q) {
    const v4f fa = *reinterpret_cast<const v4f*>(pa + 4 * q);
    const v4f fb = *reinterpret_cast<const v4f*>(pb + 4 * q);
#pragma unroll
    for (int t = 0; t < 4; ++t) {
      ba[4 * q + t] = (_Float16)fa[t];
      bb[4 * q + t] = (_Float16)fb[t];
    }
  }

  // ---- pack via WMMA ------------------------------------------------------
  const v8f c0 = {};
  const v8f cA = __builtin_amdgcn_wmma_f32_16x16x32_f16(
      false, w, false, ba, (short)0, c0, false, false);
  const v8f cB = __builtin_amdgcn_wmma_f32_16x16x32_f16(
      false, w, false, bb, (short)0, c0, false, false);

  // Accumulator layout: VGPR0 = M=0 (hi) / M=8 (hi), VGPR1 = M=1 (lo) / M=9
  // (lo); N = lane&15 = this lane's batch row.  Values are exact integers.
  const uint32_t ua = ((uint32_t)cA[0] << 16) | (uint32_t)cA[1];
  const uint32_t ub = ((uint32_t)cB[0] << 16) | (uint32_t)cB[1];

  // ---- the actual FP32 add ------------------------------------------------
  const float r = __uint_as_float(ua) + __uint_as_float(ub);
  uint32_t rb = __float_as_uint(r);
  if (((rb & 0x7F800000u) == 0x7F800000u) && (rb & 0x007FFFFFu))
    rb = 0x7FFFFFFFu;  // circuit's canonical NaN: 0 11111111 111...1

  // ---- unpack: this lane writes bits g*16 .. g*16+15 of its row -----------
  float* po = Out + rowBase;
#pragma unroll
  for (int q = 0; q < 4; ++q) {
    v4f o;
#pragma unroll
    for (int t = 0; t < 4; ++t) {
      const int j = g * 16 + q * 4 + t;          // big-endian bit index
      o[t] = (float)((rb >> (31 - j)) & 1u);
    }
    *reinterpret_cast<v4f*>(po + 4 * q) = o;
  }
}

extern "C" void kernel_launch(void* const* d_in, const int* in_sizes, int n_in,
                              void* d_out, int out_size, void* d_ws, size_t ws_size,
                              hipStream_t stream) {
  const float* A = (const float*)d_in[0];
  const float* B = (const float*)d_in[1];
  float* Out = (float*)d_out;

  const int rows   = in_sizes[0] / 32;   // BATCH (524288)
  const int ntiles = rows / 16;          // 16 rows per wave (rows % 16 == 0)
  const int wavesPerBlock = 8;           // 256 threads = 8 wave32
  const int blocks = (ntiles + wavesPerBlock - 1) / wavesPerBlock;

  fp32adder_wmma_kernel<<<blocks, 256, 0, stream>>>(A, B, Out, ntiles);
}
